// Depth_up_module_91250875171469
// MI455X (gfx1250) — matrix operand order repaired and checked by reading, not compiled, not run on hardware
//
#include <hip/hip_runtime.h>

typedef __attribute__((ext_vector_type(16))) __bf16 v16bf;
typedef __attribute__((ext_vector_type(8)))  float  v8f;
typedef __attribute__((ext_vector_type(2)))  __bf16 bf16x2;

#define N_          4
#define C_          8
#define H_          512
#define W_          640
#define TILE_W      128
#define TILES_X     5                 // 640 / 128
#define HALO_W      130
#define HALO_PITCH  132               // u16 elems per halo row
#define PATCH_PITCH 37                // u32 (bf16-pairs) per pixel column: 36 pairs + 1 pad
#define MASK_STRIDE 37
#define DEPTH_PITCH 132

union AFrag { uint4 q[2]; unsigned int u[8]; v16bf v; };
union BFrag { unsigned int u[8]; v16bf v; };

// hardware f32->bf16 (RNE); compiler picks the legal gfx1250 lowering
static __device__ __forceinline__ unsigned int pk2(float lo, float hi) {
  bf16x2 t;
  t.x = (__bf16)lo;
  t.y = (__bf16)hi;
  return __builtin_bit_cast(unsigned int, t);
}
static __device__ __forceinline__ float relu(float x) {
  return __builtin_amdgcn_fmed3f(x, 0.0f, 3.0e38f);   // v_med3_f32
}

// ---------------------------------------------------------------------------
// Prep kernel: grid-constant bf16 A-fragments (WMMA 16-bit A layout).
// Biases folded as an extra K column; the 0.25 mask scale folded into conv2.
//   ws[0   .. 767]: A1, (chunk*32 + lane)*8 + j
//   ws[768 ..1535]: A2, (m*32     + lane)*8 + j
// ---------------------------------------------------------------------------
__global__ __launch_bounds__(256)
void prep_frags(const float* __restrict__ w1, const float* __restrict__ cb1,
                const float* __restrict__ w2, const float* __restrict__ cb2,
                unsigned int* __restrict__ ws) {
  const int t = threadIdx.x;
  if (t < 96) {                       // A1: 16x96 bf16 (rows 0..7 = W1, col 72 = bias)
    const int lane = t & 31, chunk = t >> 5;
    const int half = lane >> 4, pcol = lane & 15;
    for (int j = 0; j < 8; ++j) {
      int ka = chunk * 32 + (half << 4) + (j << 1);
      float a0 = 0.f, a1 = 0.f;
      if (pcol < 8) {
        a0 = (ka     < 72) ? w1[pcol * 72 + ka]     : ((ka     == 72) ? cb1[pcol] : 0.f);
        a1 = (ka + 1 < 72) ? w1[pcol * 72 + ka + 1] : ((ka + 1 == 72) ? cb1[pcol] : 0.f);
      }
      ws[t * 8 + j] = pk2(a0, a1);
    }
  } else if (t < 192) {               // A2: 48x32 bf16, scaled by 0.25 (mask scale)
    const int u = t - 96;
    const int lane = u & 31, m = u >> 5;
    const int half = lane >> 4, pcol = lane & 15;
    const int ch = m * 16 + pcol;
    for (int j = 0; j < 8; ++j) {
      float a0 = 0.f, a1 = 0.f;
      if (ch < 36) {
        if (half == 0 && j < 4) {
          a0 = 0.25f * w2[ch * 8 + 2 * j];
          a1 = 0.25f * w2[ch * 8 + 2 * j + 1];
        } else if (half == 0 && j == 4) {
          a0 = 0.25f * cb2[ch];       // K=8 -> bias column
        }
      }
      ws[768 + u * 8 + j] = pk2(a0, a1);
    }
  }
}

// im2col expansion: one bf16 pair (k0=2*jp, k1=2*jp+1); index math folds to
// compile-time constants (jp is constexpr).
#define EXPAND_PAIR(jp_)                                                         \
  {                                                                              \
    constexpr int jp = (jp_);                                                    \
    constexpr int k0 = 2 * jp, k1 = k0 + 1;                                      \
    unsigned int lo = hp[(k0 / 9) * (3 * HALO_PITCH) + ((k0 % 9) / 3) * HALO_PITCH + (k0 % 9) % 3]; \
    unsigned int hi = hp[(k1 / 9) * (3 * HALO_PITCH) + ((k1 % 9) / 3) * HALO_PITCH + (k1 % 9) % 3]; \
    pp[jp] = lo | (hi << 16);                                                    \
  }

__global__ __launch_bounds__(256)
void depth_up_fused(const float* __restrict__ depth,
                    const float* __restrict__ cost,
                    const unsigned int* __restrict__ ws,
                    float* __restrict__ out) {
  // halo/depth layouts are shifted +3 elements so the interior float4 runs
  // store 8B/16B-aligned in LDS; readers shift their base by +3.
  __shared__ unsigned short lds_halo[24 * HALO_PITCH + 8];        // bf16 cost halo
  __shared__ unsigned int   lds_patch[TILE_W * PATCH_PITCH + 16]; // im2col bf16 pairs
  __shared__ float          lds_mask[TILE_W * MASK_STRIDE];
  __shared__ float          lds_depth[3 * DEPTH_PITCH + 8];

  const int tid = threadIdx.x;
  const int bid = blockIdx.x;
  const int tx = bid % TILES_X;
  const int h  = (bid / TILES_X) % H_;
  const int n  = bid / (TILES_X * H_);
  const int w0 = tx * TILE_W;

  // ---- Phase 1a: cost halo interior (24 rows x cols 1..128) as float4 ----
  // 768 float4 jobs: rr = job>>5 (row = i*3+r), q = job&31 (c = 1+4q)
  #pragma unroll
  for (int it = 0; it < 3; ++it) {
    int job = tid + it * 256;
    int rr = job >> 5, q = job & 31;
    int i = (rr * 11) >> 5;           // rr / 3 for rr < 24
    int r = rr - 3 * i;
    int gh = h + r - 1;
    float4 v = {0.f, 0.f, 0.f, 0.f};
    if ((unsigned)gh < (unsigned)H_)
      v = *(const float4*)&cost[((size_t)(n * C_ + i) * H_ + gh) * W_ + w0 + 4 * q];
    uint2 pkd; pkd.x = pk2(v.x, v.y); pkd.y = pk2(v.z, v.w);
    // element idx = rr*132 + (1+4q) + 3  -> byte rr*264 + 8q + 8 (8B aligned)
    *(uint2*)((char*)lds_halo + rr * (HALO_PITCH * 2) + 8 * q + 8) = pkd;
  }
  // ---- Phase 1a': halo edge columns (c=0 and c=129), 48 elements ----
  if (tid < 48) {
    int rr = tid >> 1, side = tid & 1;
    int i = (rr * 11) >> 5, r = rr - 3 * i;
    int gh = h + r - 1;
    int c  = side ? 129 : 0;
    int gw = w0 + c - 1;
    float v = 0.f;
    if ((unsigned)gh < (unsigned)H_ && (unsigned)gw < (unsigned)W_)
      v = cost[((size_t)(n * C_ + i) * H_ + gh) * W_ + gw];
    lds_halo[rr * HALO_PITCH + c + 3] = (unsigned short)(pk2(v, 0.f) & 0xFFFFu);
  }
  // ---- Phase 1b: depth halo (3 rows); interior float4 + 6 edge elements ----
  if (tid < 96) {
    int rr = tid >> 5, q = tid & 31;
    int gh = h + rr - 1;
    float4 v = {0.f, 0.f, 0.f, 0.f};
    if ((unsigned)gh < (unsigned)H_)
      v = *(const float4*)&depth[((size_t)n * H_ + gh) * W_ + w0 + 4 * q];
    // idx = rr*132 + (1+4q) + 3 -> byte rr*528 + 16q + 16 (16B aligned)
    *(float4*)((char*)lds_depth + rr * (DEPTH_PITCH * 4) + 16 * q + 16) = v;
  } else if (tid < 102) {
    int e = tid - 96;
    int rr = e >> 1, side = e & 1;
    int gh = h + rr - 1;
    int c  = side ? 129 : 0;
    int gw = w0 + c - 1;
    float v = 0.f;
    if ((unsigned)gh < (unsigned)H_ && (unsigned)gw < (unsigned)W_)
      v = depth[((size_t)n * H_ + gh) * W_ + gw];
    lds_depth[rr * DEPTH_PITCH + c + 3] = v;
  }
  __syncthreads();

  // ---- Phase 1c: im2col expansion; [pixel][k] packed bf16 pairs ----
  {
    const int pe = tid & 127;
    const unsigned short* hp = lds_halo + pe + 3;
    unsigned int* pp = lds_patch + pe * PATCH_PITCH;
    if (tid < 128) {   // wave-uniform split: waves 0-3 -> pairs 0..17
      EXPAND_PAIR(0)  EXPAND_PAIR(1)  EXPAND_PAIR(2)  EXPAND_PAIR(3)
      EXPAND_PAIR(4)  EXPAND_PAIR(5)  EXPAND_PAIR(6)  EXPAND_PAIR(7)
      EXPAND_PAIR(8)  EXPAND_PAIR(9)  EXPAND_PAIR(10) EXPAND_PAIR(11)
      EXPAND_PAIR(12) EXPAND_PAIR(13) EXPAND_PAIR(14) EXPAND_PAIR(15)
      EXPAND_PAIR(16) EXPAND_PAIR(17)
    } else {           // waves 4-7 -> pairs 18..35
      EXPAND_PAIR(18) EXPAND_PAIR(19) EXPAND_PAIR(20) EXPAND_PAIR(21)
      EXPAND_PAIR(22) EXPAND_PAIR(23) EXPAND_PAIR(24) EXPAND_PAIR(25)
      EXPAND_PAIR(26) EXPAND_PAIR(27) EXPAND_PAIR(28) EXPAND_PAIR(29)
      EXPAND_PAIR(30) EXPAND_PAIR(31) EXPAND_PAIR(32) EXPAND_PAIR(33)
      EXPAND_PAIR(34) EXPAND_PAIR(35)
    }
  }
  __syncthreads();

  // ---- Phase 2: WMMA conv1 (8->8, 3x3, relu) + conv2 (8->36, 1x1) ----
  const int lane = tid & 31, wave = tid >> 5;
  const int half = lane >> 4, pcol = lane & 15;
  const int p_local = wave * 16 + pcol;

  const unsigned int* bp =
      (const unsigned int*)((const char*)lds_patch + p_local * (PATCH_PITCH * 4) + half * 32);
  const uint4* wsq = (const uint4*)ws;

  v8f acc1 = {0.f,0.f,0.f,0.f,0.f,0.f,0.f,0.f};
  #pragma unroll
  for (int chunk = 0; chunk < 3; ++chunk) {
    AFrag a;
    a.q[0] = wsq[(chunk * 32 + lane) * 2];
    a.q[1] = wsq[(chunk * 32 + lane) * 2 + 1];
    BFrag b;
    if (chunk < 2) {
      #pragma unroll
      for (int j = 0; j < 8; ++j) b.u[j] = bp[chunk * 16 + j];
    } else {
      #pragma unroll
      for (int j = 0; j < 4; ++j) { unsigned int v = bp[32 + j]; b.u[j] = half ? 0u : v; }
      b.u[4] = half ? 0u : 0x00003f80u;   // K=72: bf16 1.0 -> bias column
      b.u[5] = 0u; b.u[6] = 0u; b.u[7] = 0u;
    }
    acc1 = __builtin_amdgcn_wmma_f32_16x16x32_bf16(
        false, a.v, false, b.v, (short)0, acc1, false, false);
  }

  // relu (bias folded); accumulator layout == conv2 B layout after packing
  BFrag b2;
  #pragma unroll
  for (int j = 0; j < 4; ++j)
    b2.u[j] = pk2(relu(acc1[2 * j]), relu(acc1[2 * j + 1]));  // lanes>=16: zeros
  b2.u[4] = half ? 0u : 0x00003f80u;    // K=8: bf16 1.0 -> bias column
  b2.u[5] = 0u; b2.u[6] = 0u; b2.u[7] = 0u;

  #pragma unroll
  for (int m = 0; m < 3; ++m) {
    AFrag a2;
    a2.q[0] = wsq[192 + (m * 32 + lane) * 2];
    a2.q[1] = wsq[192 + (m * 32 + lane) * 2 + 1];
    v8f z = {0.f,0.f,0.f,0.f,0.f,0.f,0.f,0.f};
    v8f c2 = __builtin_amdgcn_wmma_f32_16x16x32_bf16(
        false, a2.v, false, b2.v, (short)0, z, false, false);
    float* mp = lds_mask + p_local * MASK_STRIDE + m * 16 + half * 8;
    if (m < 2) {
      #pragma unroll
      for (int r = 0; r < 8; ++r) mp[r] = c2[r];     // 0.25 already folded
    } else if (half == 0) {                          // channels 32..35 only
      #pragma unroll
      for (int r = 0; r < 4; ++r) mp[r] = c2[r];
    }
  }
  __syncthreads();

  // ---- Phase 3: softmax over 9 taps + convex upsample (2 sub-rows x 128 px) ----
  const int p     = tid & 127;
  const int a_sub = tid >> 7;
  const float* dpb = lds_depth + p + 3;
  float d[9];
  #pragma unroll
  for (int t = 0; t < 9; ++t) d[t] = dpb[(t / 3) * DEPTH_PITCH + (t % 3)];

  const float* mpb = lds_mask + p * MASK_STRIDE + a_sub * 2;
  float2 res;
  #pragma unroll
  for (int bs = 0; bs < 2; ++bs) {
    float s = 0.f, acc = 0.f;
    #pragma unroll
    for (int k = 0; k < 9; ++k) {
      float e = __expf(mpb[k * 4 + bs]);   // logits are O(1): no max-sub needed
      s += e;
      acc = fmaf(e, d[k], acc);
    }
    float v = acc * __builtin_amdgcn_rcpf(s);
    if (bs == 0) res.x = v; else res.y = v;
  }
  float2* o2 = (float2*)out;
  o2[((size_t)n * 1024 + 2 * h + a_sub) * 640 + (w0 + p)] = res;
}

extern "C" void kernel_launch(void* const* d_in, const int* in_sizes, int n_in,
                              void* d_out, int out_size, void* d_ws, size_t ws_size,
                              hipStream_t stream) {
  (void)in_sizes; (void)n_in; (void)out_size; (void)ws_size;
  const float* depth = (const float*)d_in[0];
  const float* cost  = (const float*)d_in[1];
  const float* w1    = (const float*)d_in[2];
  const float* b1    = (const float*)d_in[3];
  const float* w2    = (const float*)d_in[4];
  const float* b2    = (const float*)d_in[5];
  float* out = (float*)d_out;
  unsigned int* ws = (unsigned int*)d_ws;     // needs 1536 u32 = 6 KB

  prep_frags<<<1, 256, 0, stream>>>(w1, b1, w2, b2, ws);
  depth_up_fused<<<N_ * H_ * TILES_X, 256, 0, stream>>>(depth, cost, ws, out);
}
